// SingleLayerLstm_76682346103187
// MI455X (gfx1250) — matrix operand order repaired and checked by reading, not compiled, not run on hardware
//
#include <hip/hip_runtime.h>
#include <stdint.h>

#define T_   1024
#define B_   64
#define NIN_ 512
#define H_   512
#define G_   (4 * H_)

typedef __attribute__((ext_vector_type(16))) __bf16   v16bf;
typedef __attribute__((ext_vector_type(8)))  float    v8f;
typedef __attribute__((ext_vector_type(4)))  uint32_t v4u;

union Frag32B { v16bf v; v4u q[2]; };

__device__ __forceinline__ uint16_t f32_to_bf16(float f) {
  uint32_t u = __builtin_bit_cast(uint32_t, f);
  u += 0x7FFFu + ((u >> 16) & 1u);          // round-to-nearest-even
  return (uint16_t)(u >> 16);
}

__device__ __forceinline__ float sigmoidf_(float x) {
  return 1.0f / (1.0f + __expf(-x));
}

// WGP-scope prefetch (scope field omitted -> scope 0): pulls the line into
// ALL cache levels including WGP$, unlike __builtin_prefetch which lowers to
// scope:SCOPE_SE (GL2-only, useless for already-L2-resident weights).
__device__ __forceinline__ void prefetch_wgp(const void* p) {
  asm volatile("global_prefetch_b8 %0, off" :: "v"((unsigned long long)(uintptr_t)p));
}

// ---------------- prep kernels ----------------

__global__ void k_f32_to_bf16(const float* __restrict__ in, uint16_t* __restrict__ out, int n) {
  for (int i = blockIdx.x * blockDim.x + threadIdx.x; i < n; i += gridDim.x * blockDim.x)
    out[i] = f32_to_bf16(in[i]);
}

// WT[g*K + k] = (bf16) W[k*G + g]   (make weights N-major for the WMMA B operand)
__global__ void k_transpose_to_bf16(const float* __restrict__ W, uint16_t* __restrict__ WT,
                                    int K, int G) {
  int n = K * G;
  for (int i = blockIdx.x * blockDim.x + threadIdx.x; i < n; i += gridDim.x * blockDim.x) {
    int g = i / K, k = i - g * K;
    WT[i] = f32_to_bf16(W[(size_t)k * G + g]);
  }
}

__global__ void k_copy_f32(const float* __restrict__ in, float* __restrict__ out, int n) {
  for (int i = blockIdx.x * blockDim.x + threadIdx.x; i < n; i += gridDim.x * blockDim.x)
    out[i] = in[i];
}

// ---------------- fused LSTM step ----------------
// z = [X_t ; y_prev] @ [Wf ; Wr] + b ; gates ; c ; y   as ONE K=1024 GEMM.
// grid = 32 blocks (h-tile j), 64 threads = 2 waves. Wave = 32 batch rows
// (two 16x16 M-tiles) x 4 gate tiles. Software pipeline: A operands 2-deep,
// B (weight) operands 3-deep register buffers; amdgpu_waves_per_eu(1,1) pins
// the occupancy TARGET to 1 wave/EU so the ~220-VGPR pipeline is not collapsed
// by the register-pressure scheduler. B lines are additionally prefetched
// 3 iterations ahead into WGP$ (no VGPR, no LOADcnt cost).
__global__ __launch_bounds__(64) __attribute__((amdgpu_waves_per_eu(1, 1)))
void k_lstm_step(
    const uint16_t* __restrict__ Xt,   // B x NIN  bf16 row-major
    const uint16_t* __restrict__ WfT,  // G x NIN  bf16 (Wf^T)
    const uint16_t* __restrict__ WrT,  // G x H    bf16 (Wr^T)
    const float*    __restrict__ bias, // G
    const float*    __restrict__ mask, // B   (i[t,:])
    const uint16_t* __restrict__ y_in, // B x H bf16
    const float*    __restrict__ c_prev, // B x H f32
    float*          __restrict__ Yt,   // B x H f32
    float*          __restrict__ Ct,   // B x H f32
    uint16_t*       __restrict__ y_out)// B x H bf16 (next-step carry)
{
  const int lane = threadIdx.x & 31;
  const int ln   = lane & 15;        // column (N) selector / A row selector
  const int lh   = lane >> 4;        // half-wave selector
  const int wv   = threadIdx.x >> 5; // 0..1 -> which 32-row half of the batch
  const int j    = blockIdx.x;       // 0..31 h tile
  const int m0   = wv * 32;

  v8f acc[2][4];
  #pragma unroll
  for (int s = 0; s < 4; ++s) {
    const float bv = bias[s * H_ + j * 16 + ln];   // bias depends only on column
    #pragma unroll
    for (int r = 0; r < 8; ++r) { acc[0][s][r] = bv; acc[1][s][r] = bv; }
  }

  // Row-base pointers for the concatenated [X_t ; y_prev] A operand and the
  // concatenated [Wf^T ; Wr^T] B operand.
  const uint16_t* Af[2] = { Xt + (size_t)(m0 + ln) * NIN_,
                            Xt + (size_t)(m0 + 16 + ln) * NIN_ };
  const uint16_t* Ar[2] = { y_in + (size_t)(m0 + ln) * H_,
                            y_in + (size_t)(m0 + 16 + ln) * H_ };
  const uint16_t* Bf[4];
  const uint16_t* Br[4];
  #pragma unroll
  for (int s = 0; s < 4; ++s) {
    Bf[s] = WfT + (size_t)(s * H_ + j * 16 + ln) * NIN_;
    Br[s] = WrT + (size_t)(s * H_ + j * 16 + ln) * H_;
  }

  Frag32B a[2][2];   // 2-deep A pipeline (L2-hot activations)
  Frag32B b[3][4];   // 3-deep B pipeline (streaming weights)

  auto loadA = [&](int buf, int k) {
    #pragma unroll
    for (int m = 0; m < 2; ++m) {
      const uint16_t* p = (k < NIN_) ? (Af[m] + k) : (Ar[m] + (k - NIN_));
      a[buf][m].q[0] = *(const v4u*)(p + lh * 8);
      a[buf][m].q[1] = *(const v4u*)(p + 16 + lh * 8);
    }
  };
  auto loadB = [&](int buf, int k) {
    #pragma unroll
    for (int s = 0; s < 4; ++s) {
      const uint16_t* p = (k < NIN_) ? (Bf[s] + k) : (Br[s] + (k - NIN_));
      b[buf][s].q[0] = *(const v4u*)(p + lh * 8);
      b[buf][s].q[1] = *(const v4u*)(p + 16 + lh * 8);
    }
  };
  auto prefetchB = [&](int k) {
    #pragma unroll
    for (int s = 0; s < 4; ++s) {
      const uint16_t* p = (k < NIN_) ? (Bf[s] + k) : (Br[s] + (k - NIN_));
      prefetch_wgp(p + lh * 16);
    }
  };

  constexpr int KT = NIN_ + H_;      // 1024
  loadA(0, 0);
  loadB(0, 0);
  loadB(1, 32);
  prefetchB(64);
  #pragma unroll
  for (int k = 0, it = 0; k < KT; k += 32, ++it) {
    // Issue soonest-needed first so the in-order LOADcnt drain preserves
    // B's 2-iteration lookahead.
    if (k + 32 < KT) loadA((it + 1) & 1, k + 32);
    if (k + 64 < KT) loadB((it + 2) % 3, k + 64);
    if (k + 96 < KT) prefetchB(k + 96);
    const int c2 = it & 1;
    const int c3 = it % 3;
    #pragma unroll
    for (int s = 0; s < 4; ++s) {
      acc[0][s] = __builtin_amdgcn_wmma_f32_16x16x32_bf16(
          false, a[c2][0].v, false, b[c3][s].v, (short)0, acc[0][s], false, false);
      acc[1][s] = __builtin_amdgcn_wmma_f32_16x16x32_bf16(
          false, a[c2][1].v, false, b[c3][s].v, (short)0, acc[1][s], false, false);
    }
  }

  // ---- gate epilogue: C/D layout lane = col, VGPR r -> row + r + 8*lh ----
  const int hcol = j * 16 + ln;
  #pragma unroll
  for (int mt = 0; mt < 2; ++mt) {
    #pragma unroll
    for (int r = 0; r < 8; ++r) {
      const int b_   = m0 + mt * 16 + r + lh * 8;
      const float cell = tanhf(acc[mt][0][r]);
      const float ig   = sigmoidf_(acc[mt][1][r]);
      const float fg   = sigmoidf_(acc[mt][2][r]);
      const float og   = sigmoidf_(acc[mt][3][r]);
      const float mm   = mask[b_];
      const float cp   = c_prev[(size_t)b_ * H_ + hcol];
      const float c    = (cell * ig + cp * fg) * mm + cp * (1.0f - mm);
      const float y    = tanhf(c) * og * mm;
      Ct[(size_t)b_ * H_ + hcol]    = c;
      Yt[(size_t)b_ * H_ + hcol]    = y;
      y_out[(size_t)b_ * H_ + hcol] = f32_to_bf16(y);
    }
  }
}

// ---------------- host launcher ----------------

extern "C" void kernel_launch(void* const* d_in, const int* in_sizes, int n_in,
                              void* d_out, int out_size, void* d_ws, size_t ws_size,
                              hipStream_t stream) {
  (void)in_sizes; (void)n_in; (void)out_size; (void)ws_size;

  const float* X  = (const float*)d_in[0];   // T x B x NIN
  const float* im = (const float*)d_in[1];   // T x B
  const float* h0 = (const float*)d_in[2];   // B x H
  const float* c0 = (const float*)d_in[3];   // B x H
  const float* Wf = (const float*)d_in[4];   // NIN x G
  const float* Wr = (const float*)d_in[5];   // H x G
  const float* bf = (const float*)d_in[6];   // G

  float* Y = (float*)d_out;                      // T x B x H
  float* C = Y + (size_t)T_ * B_ * H_;           // T x B x H
  float* D = C + (size_t)T_ * B_ * H_;           // B x H (final c)

  // workspace layout (bf16 = uint16_t), ~68 MB total
  uint16_t* Xbf = (uint16_t*)d_ws;                       // T*B*NIN
  uint16_t* WfT = Xbf + (size_t)T_ * B_ * NIN_;          // G*NIN
  uint16_t* WrT = WfT + (size_t)G_ * NIN_;               // G*H
  uint16_t* yb0 = WrT + (size_t)G_ * H_;                 // B*H ping
  uint16_t* yb1 = yb0 + (size_t)B_ * H_;                 // B*H pong

  // prep: convert X, transpose+convert weights, seed y carry with h0
  k_f32_to_bf16<<<8192, 256, 0, stream>>>(X, Xbf, T_ * B_ * NIN_);
  k_transpose_to_bf16<<<(G_ * NIN_ + 255) / 256, 256, 0, stream>>>(Wf, WfT, NIN_, G_);
  k_transpose_to_bf16<<<(G_ * H_ + 255) / 256, 256, 0, stream>>>(Wr, WrT, H_, G_);
  k_f32_to_bf16<<<(B_ * H_ + 255) / 256, 256, 0, stream>>>(h0, yb0, B_ * H_);

  // sequential scan: one fused GEMM+gate kernel per timestep (graph-captured)
  for (int t = 0; t < T_; ++t) {
    const uint16_t* yin  = (t & 1) ? yb1 : yb0;
    uint16_t*       yout = (t & 1) ? yb0 : yb1;
    const float*    cp   = (t == 0) ? c0 : (C + (size_t)(t - 1) * B_ * H_);
    k_lstm_step<<<32, 64, 0, stream>>>(
        Xbf + (size_t)t * B_ * NIN_, WfT, WrT, bf,
        im + (size_t)t * B_, yin, cp,
        Y + (size_t)t * B_ * H_, C + (size_t)t * B_ * H_, yout);
  }

  // d = final cell state = C[T-1]
  k_copy_f32<<<(B_ * H_ + 255) / 256, 256, 0, stream>>>(
      C + (size_t)(T_ - 1) * B_ * H_, D, B_ * H_);
}
